// SparseMultiHeadAttention_66434554134884
// MI455X (gfx1250) — matrix operand order repaired and checked by reading, not compiled, not run on hardware
//
#include <hip/hip_runtime.h>
#include <hip/hip_bf16.h>

typedef __attribute__((ext_vector_type(16))) __bf16 v16bf;
typedef __attribute__((ext_vector_type(8)))  __bf16 v8bf;
typedef __attribute__((ext_vector_type(8)))  float  v8f;

// async-to-LDS builtin takes pointers to int4 vectors (per hipcc diagnostic).
typedef int v4i __attribute__((vector_size(16)));
typedef __attribute__((address_space(1))) v4i as1_v4i;   // global ("__device__")
typedef __attribute__((address_space(3))) v4i as3_v4i;   // LDS

#define DEVINL __device__ __forceinline__

// Compile-safe detection of the gfx1250 async-to-LDS builtins.
#if defined(__has_builtin)
#  if __has_builtin(__builtin_amdgcn_global_load_async_to_lds_b128)
#    define USE_ASYNC_LDS 1
#  endif
#endif

DEVINL void wait_asynccnt0() {
#if defined(__has_builtin) && __has_builtin(__builtin_amdgcn_s_wait_asynccnt)
  __builtin_amdgcn_s_wait_asynccnt(0);
#else
  asm volatile("s_wait_asynccnt 0x0" ::: "memory");
#endif
}

DEVINL v8f wmma_bf16(v16bf a, v16bf b, v8f c) {
  // v_wmma_f32_16x16x32_bf16: D = A(16x32) x B(32x16) + C(16x16 f32)
  return __builtin_amdgcn_wmma_f32_16x16x32_bf16(
      /*neg_a=*/false, a, /*neg_b=*/false, b,
      /*c_mod=*/(short)0, c, /*reuse_a=*/false, /*reuse_b=*/false);
}

// A-operand (16x32) from row-major bf16 [., ld] at (row0, k0).
// ISA layout: lane = M (lane&15); half (lane>>4) selects K groups
//   elems 0..7  -> K = k0 + half*8 + e ; elems 8..15 -> K = k0 + 16 + half*8 + (e-8)
DEVINL v16bf load_a_tile(const __bf16* base, int ld, int row0, int k0) {
  const int lane = threadIdx.x & 31;
  const int m = lane & 15;
  const int half = lane >> 4;
  const __bf16* p = base + (size_t)(row0 + m) * ld + (k0 + half * 8);
  v8bf lo = *(const v8bf*)p;
  v8bf hi = *(const v8bf*)(p + 16);
  v16bf a;
#pragma unroll
  for (int i = 0; i < 8; ++i) { a[i] = lo[i]; a[i + 8] = hi[i]; }
  return a;
}

// B-operand (32x16) from row-major bf16 [K][ld] at (k0, col0).
// ISA layout: lane&15 = N column; elems 0..7 -> K = k0 + half*8 + e ; elems 8..15 -> K = k0 + 16 + half*8 + (e-8)
DEVINL v16bf load_b_tile(const __bf16* base, int ld, int k0, int col0) {
  const int lane = threadIdx.x & 31;
  const __bf16* p = base + (size_t)(k0 + (lane >> 4) * 8) * ld + col0 + (lane & 15);
  v16bf b;
#pragma unroll
  for (int i = 0; i < 8; ++i) { b[i] = p[(size_t)i * ld]; b[i + 8] = p[(size_t)(i + 16) * ld]; }
  return b;
}

// ---------------------------------------------------------------- cast f32->bf16
__global__ void __launch_bounds__(256)
cast_kernel(const float* __restrict__ src, __bf16* __restrict__ dst, int n) {
  int i = blockIdx.x * blockDim.x + threadIdx.x;
  if (i < n) dst[i] = (__bf16)src[i];
}

// ------------------------------------------------- transpose + cast f32 -> bf16
// src: [rows][cols] f32 row-major; dst: [cols][rows] bf16 row-major.
__global__ void __launch_bounds__(256)
transpose_cast_kernel(const float* __restrict__ src, __bf16* __restrict__ dst,
                      int rows, int cols) {
  __shared__ __bf16 tile[32][40];
  const int c0 = blockIdx.x * 32;
  const int r0 = blockIdx.y * 32;
  const int tx = threadIdx.x;   // 0..31
  const int ty = threadIdx.y;   // 0..7
#pragma unroll
  for (int i = 0; i < 32; i += 8)
    tile[ty + i][tx] = (__bf16)src[(size_t)(r0 + ty + i) * cols + (c0 + tx)];
  __syncthreads();
#pragma unroll
  for (int i = 0; i < 32; i += 8)
    dst[(size_t)(c0 + ty + i) * rows + (r0 + tx)] = tile[tx][ty + i];
}

// ----------------------------------------------------------- GEMM + bias (f32 out)
// Cout[M][Ncols] = A(bf16,[M][K]) * B(bf16,[K][Ncols]) + bias.
// 256 threads = 8 waves arranged 2 (row slabs) x 4 (col slabs).
// Block tile = 32 rows x 256 cols. Per k-step: the 32x32 A panel is staged
// into LDS once (async-to-LDS when available), each wave reuses its A
// operand across 4 B tiles -> 32 WMMAs per block per k-step.
__global__ void __launch_bounds__(256)
gemm_bf16_bias(const __bf16* __restrict__ A, const __bf16* __restrict__ B,
               const float* __restrict__ bias, float* __restrict__ Cout,
               int Ncols, int K) {
  __shared__ __align__(32) __bf16 ldsA[32 * 40];   // 32x32 panel, padded rows

  const int wave = threadIdx.x >> 5;
  const int lane = threadIdx.x & 31;
  const int wr = wave >> 2;          // 0..1 row slab
  const int wc = wave & 3;           // 0..3 col slab
  const int lx = lane & 15;
  const int half = lane >> 4;

  const int rowblk = blockIdx.x * 32;
  const int row0 = rowblk + wr * 16;
  const int col0 = blockIdx.y * 256 + wc * 64;

  v8f acc[4] = {};

  for (int k0 = 0; k0 < K; k0 += 32) {
    // ---- stage A panel (32 rows x 32 K) into LDS, wave 0 only ----
    if (wave == 0) {
#pragma unroll
      for (int i = 0; i < 4; ++i) {
        const int r = i * 8 + (lane >> 2);       // 0..31
        const int ch = (lane & 3) * 8;           // K element chunk (16B)
        const __bf16* g = A + (size_t)(rowblk + r) * K + k0 + ch;
        __bf16* l = &ldsA[r * 40 + ch];
#if defined(USE_ASYNC_LDS)
        __builtin_amdgcn_global_load_async_to_lds_b128(
            (as1_v4i*)g, (as3_v4i*)l, /*offset=*/0, /*cpol=*/0);
#else
        *(v8bf*)l = *(const v8bf*)g;
#endif
      }
#if defined(USE_ASYNC_LDS)
      wait_asynccnt0();
#endif
    }
    __syncthreads();

    if (k0 + 64 < K)   // warm GL2 for the B panel two steps ahead
      __builtin_prefetch(B + (size_t)(k0 + 64 + lane) * Ncols + col0, 0, 1);

    // A operand from LDS (this wave's 16-row slab)
    v16bf a;
    {
      const __bf16* pr = &ldsA[(wr * 16 + lx) * 40 + half * 8];
      v8bf lo = *(const v8bf*)pr;
      v8bf hi = *(const v8bf*)(pr + 16);
#pragma unroll
      for (int i = 0; i < 8; ++i) { a[i] = lo[i]; a[i + 8] = hi[i]; }
    }

#pragma unroll
    for (int t = 0; t < 4; ++t) {
      v16bf b = load_b_tile(B, Ncols, k0, col0 + 16 * t);
      acc[t] = wmma_bf16(a, b, acc[t]);
    }
    __syncthreads();   // before next k-step overwrites ldsA
  }

  // epilogue: bias + f32 store (C layout: VGPR j -> row j+8*half, lane&15 -> col)
#pragma unroll
  for (int t = 0; t < 4; ++t) {
    const int col = col0 + 16 * t + lx;
    const float bv = bias[col];
#pragma unroll
    for (int j = 0; j < 8; ++j)
      Cout[(size_t)(row0 + j + 8 * half) * Ncols + col] = acc[t][j] + bv;
  }
}

// ---------------------------------------- per-head RMS norm + split/pack q,k,v
// qkv f32 [N][3C]. Writes qb [N][C], vb [N][C] (row-major bf16) and
// kT [H*D][N] bf16 (k transposed -> contiguous B-operand rows for Q@K^T).
__global__ void __launch_bounds__(512)
rmsnorm_split_kernel(const float* __restrict__ qkv,
                     const float* __restrict__ qg, const float* __restrict__ kg,
                     __bf16* __restrict__ qb, __bf16* __restrict__ kT,
                     __bf16* __restrict__ vb) {
  const int n = blockIdx.x;          // token row
  const int h = threadIdx.y;         // head (one wave per head)
  const int lane = threadIdx.x;      // 0..31, each lane owns 2 of D=64 elems
  const size_t base = (size_t)n * 3072 + (size_t)h * 64;
  {  // q
    const float* p = qkv + base;
    float a = p[lane], b = p[lane + 32];
    float ss = a * a + b * b;
#pragma unroll
    for (int m = 16; m >= 1; m >>= 1) ss += __shfl_xor(ss, m, 32);
    const float inv = 8.0f / fmaxf(sqrtf(ss), 1e-12f);   // sqrt(D)/||q||
    qb[(size_t)n * 1024 + h * 64 + lane]      = (__bf16)(a * inv * qg[h * 64 + lane]);
    qb[(size_t)n * 1024 + h * 64 + lane + 32] = (__bf16)(b * inv * qg[h * 64 + lane + 32]);
  }
  {  // k (transposed store)
    const float* p = qkv + base + 1024;
    float a = p[lane], b = p[lane + 32];
    float ss = a * a + b * b;
#pragma unroll
    for (int m = 16; m >= 1; m >>= 1) ss += __shfl_xor(ss, m, 32);
    const float inv = 8.0f / fmaxf(sqrtf(ss), 1e-12f);
    kT[(size_t)(h * 64 + lane) * 4096 + n]      = (__bf16)(a * inv * kg[h * 64 + lane]);
    kT[(size_t)(h * 64 + lane + 32) * 4096 + n] = (__bf16)(b * inv * kg[h * 64 + lane + 32]);
  }
  {  // v
    const float* p = qkv + base + 2048;
    vb[(size_t)n * 1024 + h * 64 + lane]      = (__bf16)p[lane];
    vb[(size_t)n * 1024 + h * 64 + lane + 32] = (__bf16)p[lane + 32];
  }
}

// ------------------------------------------------------------ flash attention
// One wave per (16 q-rows, head). Online softmax over 32-key chunks.
// Per-row stats live as 8 scalars/lane: lanes 0-15 own rows 0-7 at column
// lane&15; lanes 16-31 own rows 8-15 (matches WMMA C-layout exactly).
__global__ void __launch_bounds__(32)
flash_attn_kernel(const __bf16* __restrict__ qb, const __bf16* __restrict__ kT,
                  const __bf16* __restrict__ vb, __bf16* __restrict__ hb) {
  const int N = 4096, C = 1024;
  const int q0 = blockIdx.x * 16;
  const int h = blockIdx.y;
  const int lane = threadIdx.x;
  const int lx = lane & 15;
  const int half = lane >> 4;

  __shared__ __align__(32) __bf16 pT[16 * 40];   // P transpose staging (C->A layout)

  // Q A-operands for K-dim D=64 (two 16x32 slabs), reused across all chunks.
  const v16bf a0 = load_a_tile(qb + h * 64, C, q0, 0);
  const v16bf a1 = load_a_tile(qb + h * 64, C, q0, 32);

  v8f acc0 = {}, acc1 = {}, acc2 = {}, acc3 = {};   // 16x64 f32 accumulator
  float mrow[8], lrow[8];
#pragma unroll
  for (int j = 0; j < 8; ++j) { mrow[j] = -3.0e38f; lrow[j] = 0.0f; }

  const __bf16* kTh = kT + (size_t)h * 64 * N;
  for (int m0 = 0; m0 < N; m0 += 32) {
    // S(16x32) = Q(16x64) @ K^T(64x32): lane&15 = key col, elems = d rows of kT.
    const __bf16* kc = kTh + (size_t)(half * 8) * N + m0 + lx;
    v16bf b00, b01, b10, b11;
#pragma unroll
    for (int i = 0; i < 8; ++i) {
      b00[i] = kc[(size_t)i * N];             b00[i + 8] = kc[(size_t)(i + 16) * N];
      b01[i] = kc[(size_t)i * N + 16];        b01[i + 8] = kc[(size_t)(i + 16) * N + 16];
      b10[i] = kc[(size_t)(i + 32) * N];      b10[i + 8] = kc[(size_t)(i + 48) * N];
      b11[i] = kc[(size_t)(i + 32) * N + 16]; b11[i + 8] = kc[(size_t)(i + 48) * N + 16];
    }

    v8f s0 = {}, s1 = {};
    s0 = wmma_bf16(a0, b00, s0);
    s0 = wmma_bf16(a1, b10, s0);
    s1 = wmma_bf16(a0, b01, s1);
    s1 = wmma_bf16(a1, b11, s1);

    // online softmax: per-row max across the 16 lanes of each half-group
    float rmax[8];
#pragma unroll
    for (int j = 0; j < 8; ++j) {
      s0[j] *= 0.125f;                       // 1/sqrt(D)
      s1[j] *= 0.125f;
      rmax[j] = fmaxf(s0[j], s1[j]);
    }
#pragma unroll
    for (int msk = 8; msk >= 1; msk >>= 1)
#pragma unroll
      for (int j = 0; j < 8; ++j)
        rmax[j] = fmaxf(rmax[j], __shfl_xor(rmax[j], msk, 16));

    float alpha[8], rsum[8];
#pragma unroll
    for (int j = 0; j < 8; ++j) {
      const float mn = fmaxf(mrow[j], rmax[j]);
      alpha[j] = __expf(mrow[j] - mn);
      s0[j] = __expf(s0[j] - mn);
      s1[j] = __expf(s1[j] - mn);
      rsum[j] = s0[j] + s1[j];
      mrow[j] = mn;
    }
#pragma unroll
    for (int msk = 8; msk >= 1; msk >>= 1)
#pragma unroll
      for (int j = 0; j < 8; ++j)
        rsum[j] += __shfl_xor(rsum[j], msk, 16);
#pragma unroll
    for (int j = 0; j < 8; ++j) {
      lrow[j] = lrow[j] * alpha[j] + rsum[j];
      acc0[j] *= alpha[j]; acc1[j] *= alpha[j];
      acc2[j] *= alpha[j]; acc3[j] *= alpha[j];
    }

    // P: C-layout -> A-layout via LDS (same wave; compiler orders via DScnt)
#pragma unroll
    for (int j = 0; j < 8; ++j) {
      const int row = j + 8 * half;
      pT[row * 40 + lx]      = (__bf16)s0[j];
      pT[row * 40 + 16 + lx] = (__bf16)s1[j];
    }
    v16bf pa;
    {
      const __bf16* pr = pT + lx * 40 + half * 8;
      v8bf lo = *(const v8bf*)pr;
      v8bf hi = *(const v8bf*)(pr + 16);
#pragma unroll
      for (int i = 0; i < 8; ++i) { pa[i] = lo[i]; pa[i + 8] = hi[i]; }
    }

    // acc(16x64) += P(16x32) @ V(32x64); lane&15 = d col of V, elems = key rows.
    const __bf16* vrow = vb + (size_t)(m0 + half * 8) * C + h * 64 + lx;
    v16bf v0, v1, v2, v3;
#pragma unroll
    for (int i = 0; i < 8; ++i) {
      v0[i] = vrow[(size_t)i * C];      v0[i + 8] = vrow[(size_t)(i + 16) * C];
      v1[i] = vrow[(size_t)i * C + 16]; v1[i + 8] = vrow[(size_t)(i + 16) * C + 16];
      v2[i] = vrow[(size_t)i * C + 32]; v2[i + 8] = vrow[(size_t)(i + 16) * C + 32];
      v3[i] = vrow[(size_t)i * C + 48]; v3[i + 8] = vrow[(size_t)(i + 16) * C + 48];
    }
    acc0 = wmma_bf16(pa, v0, acc0);
    acc1 = wmma_bf16(pa, v1, acc1);
    acc2 = wmma_bf16(pa, v2, acc2);
    acc3 = wmma_bf16(pa, v3, acc3);
  }

#pragma unroll
  for (int j = 0; j < 8; ++j) {
    const float invl = 1.0f / lrow[j];
    __bf16* hr = hb + (size_t)(q0 + j + 8 * half) * C + h * 64;
    hr[lx]      = (__bf16)(acc0[j] * invl);
    hr[lx + 16] = (__bf16)(acc1[j] * invl);
    hr[lx + 32] = (__bf16)(acc2[j] * invl);
    hr[lx + 48] = (__bf16)(acc3[j] * invl);
  }
}

// ------------------------------------------------------------------ launcher
extern "C" void kernel_launch(void* const* d_in, const int* in_sizes, int n_in,
                              void* d_out, int out_size, void* d_ws, size_t ws_size,
                              hipStream_t stream) {
  (void)in_sizes; (void)n_in; (void)out_size; (void)ws_size;
  const float* x    = (const float*)d_in[0];
  const float* Wqkv = (const float*)d_in[1];
  const float* bqkv = (const float*)d_in[2];
  const float* qg   = (const float*)d_in[3];
  const float* kg   = (const float*)d_in[4];
  const float* Wout = (const float*)d_in[5];
  const float* bout = (const float*)d_in[6];
  float* out = (float*)d_out;

  const int N = 4096, C = 1024;
  char* ws = (char*)d_ws;
  size_t off = 0;
  float*  qkv   = (float*)(ws + off);  off += (size_t)N * 3 * C * sizeof(float); // 48 MB
  __bf16* xb    = (__bf16*)(ws + off); off += (size_t)N * C * 2;                 //  8 MB
  __bf16* WqkvT = (__bf16*)(ws + off); off += (size_t)3 * C * C * 2;             //  6 MB
  __bf16* WoutT = (__bf16*)(ws + off); off += (size_t)C * C * 2;                 //  2 MB
  __bf16* qbuf  = (__bf16*)(ws + off); off += (size_t)N * C * 2;                 //  8 MB
  __bf16* kTbuf = (__bf16*)(ws + off); off += (size_t)N * C * 2;                 //  8 MB
  __bf16* vbuf  = (__bf16*)(ws + off); off += (size_t)N * C * 2;                 //  8 MB
  __bf16* hbuf  = (__bf16*)(ws + off); off += (size_t)N * C * 2;                 //  8 MB

  cast_kernel<<<(N * C + 255) / 256, 256, 0, stream>>>(x, xb, N * C);
  transpose_cast_kernel<<<dim3(C / 32, 3 * C / 32), dim3(32, 8), 0, stream>>>(
      Wqkv, WqkvT, 3 * C, C);
  transpose_cast_kernel<<<dim3(C / 32, C / 32), dim3(32, 8), 0, stream>>>(
      Wout, WoutT, C, C);
  // qkv = x @ Wqkv^T + bqkv      (block tile 32 x 256)
  gemm_bf16_bias<<<dim3(N / 32, 3 * C / 256), 256, 0, stream>>>(
      xb, WqkvT, bqkv, qkv, 3 * C, C);
  rmsnorm_split_kernel<<<N, dim3(32, 16), 0, stream>>>(
      qkv, qg, kg, qbuf, kTbuf, vbuf);
  flash_attn_kernel<<<dim3(N / 16, 16), 32, 0, stream>>>(qbuf, kTbuf, vbuf, hbuf);
  // out = h @ Wout^T + bout
  gemm_bf16_bias<<<dim3(N / 32, C / 256), 256, 0, stream>>>(
      hbuf, WoutT, bout, out, C, C);
}
